// ResidualIRT_44659069944385
// MI455X (gfx1250) — compile-verified, matrix operand order checked by reading, not run on hardware
//
#include <hip/hip_runtime.h>
#include <hip/hip_bf16.h>

typedef __attribute__((ext_vector_type(16))) _Float16 v16h;
typedef __attribute__((ext_vector_type(8)))  _Float16 v8h;
typedef __attribute__((ext_vector_type(8)))  float    v8f;

#define PN 2048
#define IN 512
#define DN 32
#define HN 128
#define F_ITEM 33

__device__ __forceinline__ float eluf(float x) {
    return x > 0.0f ? x : (__expf(x) - 1.0f);
}

// ---------------------------------------------------------------------------
// Prep 1: per-row 3-layer MLP + Wc1 projection.  blockDim = 128 (one feature
// per thread), one block per row.  Shared for the ability tower
// (in_dim=32, wc1_row_off=128, add_bias=bc1 folded in) and the item tower
// (in_dim=33, wc1_row_off=0, add_bias=nullptr).
// ---------------------------------------------------------------------------
__global__ void mlp_proj_kernel(const float* __restrict__ in, int in_dim,
                                const float* __restrict__ W1, const float* __restrict__ b1,
                                const float* __restrict__ W2, const float* __restrict__ b2,
                                const float* __restrict__ W3, const float* __restrict__ b3,
                                const float* __restrict__ Wc1, int wc1_row_off,
                                const float* __restrict__ add_bias,
                                float* __restrict__ proj) {
    __shared__ float s_in[64];
    __shared__ float sh0[HN];
    __shared__ float sh1[HN];
    const int p = blockIdx.x;
    const int t = threadIdx.x;

    if (t < in_dim) s_in[t] = in[p * in_dim + t];
    __syncthreads();

    float acc = b1[t];
    for (int k = 0; k < in_dim; ++k) acc += s_in[k] * W1[k * HN + t];
    sh0[t] = eluf(acc);
    __syncthreads();

    acc = b2[t];
    #pragma unroll 4
    for (int k = 0; k < HN; ++k) acc += sh0[k] * W2[k * HN + t];
    sh1[t] = eluf(acc);
    __syncthreads();

    acc = b3[t];
    #pragma unroll 4
    for (int k = 0; k < HN; ++k) acc += sh1[k] * W3[k * HN + t];
    __syncthreads();
    sh0[t] = acc;                 // final ha / hi (no elu)
    __syncthreads();

    float pr = 0.0f;
    #pragma unroll 4
    for (int k = 0; k < HN; ++k) pr += sh0[k] * Wc1[(wc1_row_off + k) * HN + t];
    if (add_bias) pr += add_bias[t];       // fold bc1 into proj_a
    proj[p * HN + t] = pr;
}

// ---------------------------------------------------------------------------
// Prep 2: irt[p,i] = -ability[p,:] . item_feat[i,0:32] + item_feat[i,32] + bc3
// (bc3 folded in here so the main-kernel epilogue is just part + irt.)
// ---------------------------------------------------------------------------
__global__ void irt_kernel(const float* __restrict__ ability,
                           const float* __restrict__ item_feat,
                           const float* __restrict__ bc3,
                           float* __restrict__ irt) {
    const int j = blockIdx.x * blockDim.x + threadIdx.x;   // j = p*IN + i
    if (j >= PN * IN) return;
    const int p = j >> 9;        // /512
    const int i = j & (IN - 1);
    const float* ab  = ability + p * DN;
    const float* itf = item_feat + i * F_ITEM;
    float a = 0.0f;
    #pragma unroll
    for (int k = 0; k < DN; ++k) a += ab[k] * itf[k];
    irt[j] = itf[DN] - a + bc3[0];
}

// ---------------------------------------------------------------------------
// Prep 3: Wc2 [K=128][N=128] f32 row-major  ->  Wc2^T [N][K] f16
// ---------------------------------------------------------------------------
__global__ void wc2_transpose_f16_kernel(const float* __restrict__ Wc2,
                                         _Float16* __restrict__ wc2t) {
    const int idx = blockIdx.x * blockDim.x + threadIdx.x;  // n*128 + k
    if (idx >= HN * HN) return;
    const int n = idx >> 7;
    const int k = idx & (HN - 1);
    wc2t[idx] = (_Float16)Wc2[k * HN + n];
}

// ---------------------------------------------------------------------------
// Main kernel: per wave, one (p, 16-item) tile.
//   A (16x128, f16, built on the fly) x B (128x128, Wc2^T from LDS)
//   via 4(K) x 8(N) v_wmma_f32_16x16x32_f16, then elu(+bc2), dot Wc3,
//   cross-lane reduce, + irt(+bc3), sigmoid.
// blockDim = 256 (8 waves); grid = P*(I/16)/8 = 8192 blocks.
// ---------------------------------------------------------------------------
__global__ void __launch_bounds__(256)
residual_irt_main_kernel(const float* __restrict__ proj_a,   // includes +bc1
                         const float* __restrict__ proj_i,
                         const float* __restrict__ irt,      // includes +bc3
                         const _Float16* __restrict__ wc2t,
                         const float* __restrict__ bc2,
                         const float* __restrict__ Wc3,
                         float* __restrict__ out) {
    __shared__ _Float16 sB[HN * HN];          // 32 KB: Wc2^T [n][k] in f16

    // Cooperative stage of Wc2^T into LDS (uniform, all lanes active).
    {
        const uint4* src = (const uint4*)wc2t;
        uint4* dst = (uint4*)sB;
        #pragma unroll
        for (int rep = 0; rep < (HN * HN / 8) / 256; ++rep)
            dst[rep * 256 + threadIdx.x] = src[rep * 256 + threadIdx.x];
    }
    __syncthreads();

    const int wave = threadIdx.x >> 5;
    const int lane = threadIdx.x & 31;
    const int tile = blockIdx.x * 8 + wave;   // 0 .. 65535
    const int p  = tile >> 5;                 // 32 item-tiles per person
    const int i0 = (tile & 31) * 16;

    const int r    = lane & 15;               // row within tile (item), also N col
    const int koff = (lane >> 4) * 8;         // 0 or 8 per ISA 16-bit A layout
    const int half = lane >> 4;

    const float* __restrict__ pa = proj_a + p * HN;
    const float* __restrict__ pi = proj_i + (i0 + r) * HN;

    v8f acc[8];
    #pragma unroll
    for (int nt = 0; nt < 8; ++nt) acc[nt] = (v8f)(0.0f);

    #pragma unroll
    for (int kt = 0; kt < 4; ++kt) {
        const int kb = kt * 32;

        // ---- Build A fragment: h = elu(proj_a' + proj_i) -> f16 ----
        v16h a;
        {
            const int k0 = kb + koff;          // halfs 0..7  -> K = k0..k0+7
            const int k1 = kb + 16 + koff;     // halfs 8..15 -> K = k1..k1+7
            const float4 pa0 = *(const float4*)(pa + k0);
            const float4 pa1 = *(const float4*)(pa + k0 + 4);
            const float4 pa2 = *(const float4*)(pa + k1);
            const float4 pa3 = *(const float4*)(pa + k1 + 4);
            const float4 pi0 = *(const float4*)(pi + k0);
            const float4 pi1 = *(const float4*)(pi + k0 + 4);
            const float4 pi2 = *(const float4*)(pi + k1);
            const float4 pi3 = *(const float4*)(pi + k1 + 4);
            float va[16];
            va[0]=pa0.x+pi0.x;   va[1]=pa0.y+pi0.y;
            va[2]=pa0.z+pi0.z;   va[3]=pa0.w+pi0.w;
            va[4]=pa1.x+pi1.x;   va[5]=pa1.y+pi1.y;
            va[6]=pa1.z+pi1.z;   va[7]=pa1.w+pi1.w;
            va[8]=pa2.x+pi2.x;   va[9]=pa2.y+pi2.y;
            va[10]=pa2.z+pi2.z;  va[11]=pa2.w+pi2.w;
            va[12]=pa3.x+pi3.x;  va[13]=pa3.y+pi3.y;
            va[14]=pa3.z+pi3.z;  va[15]=pa3.w+pi3.w;
            #pragma unroll
            for (int e = 0; e < 16; ++e) a[e] = (_Float16)eluf(va[e]);
        }

        // ---- 8 N-tiles: B fragment from LDS + WMMA ----
        #pragma unroll
        for (int nt = 0; nt < 8; ++nt) {
            const int n = nt * 16 + r;         // this lane's B column
            const v8h b0 = *(const v8h*)(sB + n * HN + kb + koff);
            const v8h b1 = *(const v8h*)(sB + n * HN + kb + 16 + koff);
            v16h b;
            #pragma unroll
            for (int e = 0; e < 8; ++e) { b[e] = b0[e]; b[8 + e] = b1[e]; }
            acc[nt] = __builtin_amdgcn_wmma_f32_16x16x32_f16(
                /*neg_a=*/false, a, /*neg_b=*/false, b,
                /*c_mod=*/(short)0, acc[nt],
                /*reuse_a=*/false, /*reuse_b=*/false);
        }
    }

    // ---- Epilogue: elu(acc + bc2[n]) * Wc3[n], reduce over n (lanes) ----
    float part[8];
    #pragma unroll
    for (int v = 0; v < 8; ++v) part[v] = 0.0f;

    #pragma unroll
    for (int nt = 0; nt < 8; ++nt) {
        const int n = nt * 16 + r;             // C/D layout: lane%16 == N
        const float bcn = bc2[n];
        const float w3  = Wc3[n];
        #pragma unroll
        for (int v = 0; v < 8; ++v) {
            const float x = eluf(acc[nt][v] + bcn);
            part[v] += x * w3;
        }
    }
    // Butterfly across the 16 lanes of each half (rows m = v + 8*half).
    #pragma unroll
    for (int off = 1; off < 16; off <<= 1) {
        #pragma unroll
        for (int v = 0; v < 8; ++v)
            part[v] += __shfl_xor(part[v], off, 32);
    }

    if (r == 0) {
        const int base = p * IN + i0 + 8 * half;   // 8 contiguous outputs
        const float4 ir0 = *(const float4*)(irt + base);
        const float4 ir1 = *(const float4*)(irt + base + 4);
        float4 o0, o1;
        o0.x = 1.0f / (1.0f + __expf(-(part[0] + ir0.x)));
        o0.y = 1.0f / (1.0f + __expf(-(part[1] + ir0.y)));
        o0.z = 1.0f / (1.0f + __expf(-(part[2] + ir0.z)));
        o0.w = 1.0f / (1.0f + __expf(-(part[3] + ir0.w)));
        o1.x = 1.0f / (1.0f + __expf(-(part[4] + ir1.x)));
        o1.y = 1.0f / (1.0f + __expf(-(part[5] + ir1.y)));
        o1.z = 1.0f / (1.0f + __expf(-(part[6] + ir1.z)));
        o1.w = 1.0f / (1.0f + __expf(-(part[7] + ir1.w)));
        *(float4*)(out + base)     = o0;
        *(float4*)(out + base + 4) = o1;
    }
}

// ---------------------------------------------------------------------------
extern "C" void kernel_launch(void* const* d_in, const int* in_sizes, int n_in,
                              void* d_out, int out_size, void* d_ws, size_t ws_size,
                              hipStream_t stream) {
    const float* ability   = (const float*)d_in[0];
    const float* item_feat = (const float*)d_in[1];
    const float* Wa1 = (const float*)d_in[2];
    const float* ba1 = (const float*)d_in[3];
    const float* Wa2 = (const float*)d_in[4];
    const float* ba2 = (const float*)d_in[5];
    const float* Wa3 = (const float*)d_in[6];
    const float* ba3 = (const float*)d_in[7];
    const float* Wi1 = (const float*)d_in[8];
    const float* bi1 = (const float*)d_in[9];
    const float* Wi2 = (const float*)d_in[10];
    const float* bi2 = (const float*)d_in[11];
    const float* Wi3 = (const float*)d_in[12];
    const float* bi3 = (const float*)d_in[13];
    const float* Wc1 = (const float*)d_in[14];
    const float* bc1 = (const float*)d_in[15];
    const float* Wc2 = (const float*)d_in[16];
    const float* bc2 = (const float*)d_in[17];
    const float* Wc3 = (const float*)d_in[18];
    const float* bc3 = (const float*)d_in[19];
    float* out = (float*)d_out;

    // Workspace layout
    char* ws = (char*)d_ws;
    float*    proj_a = (float*)(ws);                              // 2048*128 f32 = 1 MB
    float*    proj_i = (float*)(ws + (size_t)PN * HN * 4);        // 512*128 f32  = 256 KB
    float*    irt    = (float*)(ws + (size_t)(PN + IN) * HN * 4); // 2048*512 f32 = 4 MB
    _Float16* wc2t   = (_Float16*)(ws + (size_t)(PN + IN) * HN * 4
                                      + (size_t)PN * IN * 4);     // 128*128 f16 = 32 KB

    // 1) MLP towers -> proj_a (+bc1) / proj_i
    mlp_proj_kernel<<<PN, HN, 0, stream>>>(ability, DN, Wa1, ba1, Wa2, ba2,
                                           Wa3, ba3, Wc1, HN, bc1, proj_a);
    mlp_proj_kernel<<<IN, HN, 0, stream>>>(item_feat, F_ITEM, Wi1, bi1, Wi2, bi2,
                                           Wi3, bi3, Wc1, 0, nullptr, proj_i);
    // 2) IRT logits (+bc3)
    irt_kernel<<<(PN * IN) / 256, 256, 0, stream>>>(ability, item_feat, bc3, irt);
    // 3) Wc2^T in f16
    wc2_transpose_f16_kernel<<<(HN * HN) / 256, 256, 0, stream>>>(Wc2, wc2t);
    // 4) Main pairwise WMMA kernel
    residual_irt_main_kernel<<<(PN * (IN / 16)) / 8, 256, 0, stream>>>(
        proj_a, proj_i, irt, wc2t, bc2, Wc3, out);
}
